// FeedFoward_52372831207465
// MI455X (gfx1250) — 8-bit matrix operand order repaired and checked by reading, compiled, not run on hardware
//
#include <hip/hip_runtime.h>

typedef int   v16i __attribute__((ext_vector_type(16)));
typedef int   v4i  __attribute__((ext_vector_type(4)));
typedef float v8f  __attribute__((ext_vector_type(8)));

#define N_EMBD 96
#define HIDDEN 384
#define NT1 24
#define NT2 6
#define KT2 3

#define W1A_N   (NT1 * 16 * 32)
#define W2A_N   (KT2 * NT2 * 16 * 32)
#define W1A_OFF 0
#define W2A_OFF (W1A_OFF + W1A_N)
#define B1_OFF  (W2A_OFF + W2A_N)
#define B2_OFF  (B1_OFF + HIDDEN)

#if __has_builtin(__builtin_amdgcn_wmma_f32_16x16x128_bf8_bf8)
#define WMMA128(A, B, C) __builtin_amdgcn_wmma_f32_16x16x128_bf8_bf8((A), (B), (short)0, (C), false, false)
#else
#define WMMA128(A, B, C) __builtin_amdgcn_wmma_f32_16x16x128_fp8_fp8((A), (B), (short)0, (C), false, false)
#endif

#if __has_builtin(__builtin_amdgcn_cvt_pk_bf8_f32)
#define HAVE_PK_BF8 1
#endif
#if __has_builtin(__builtin_amdgcn_global_load_async_to_lds_b128)
#define HAVE_ASYNC 1
#endif

__device__ __forceinline__ unsigned bf8_byte(float f) {
  unsigned short h = __builtin_bit_cast(unsigned short, (_Float16)f);
  return (((unsigned)h + 0x7Fu + ((h >> 8) & 1u)) >> 8) & 0xFFu;
}
__device__ __forceinline__ float bf8_decode(unsigned b) {
  unsigned short h = (unsigned short)(b << 8);
  return (float)__builtin_bit_cast(_Float16, h);
}
__device__ __forceinline__ unsigned pack4_bf8(float a, float b, float c, float d) {
#if defined(HAVE_PK_BF8)
  int v = __builtin_amdgcn_cvt_pk_bf8_f32(a, b, 0, false);
  v = __builtin_amdgcn_cvt_pk_bf8_f32(c, d, v, true);
  return (unsigned)v;
#else
  return bf8_byte(a) | (bf8_byte(b) << 8) | (bf8_byte(c) << 16) | (bf8_byte(d) << 24);
#endif
}
__device__ __forceinline__ float f16r(float x) { return (float)(_Float16)x; }

__device__ __forceinline__ int a_k0(int v, int hi) {
  return ((v >> 2) * 32) + hi * 16 + ((v & 3) * 4);
}
__device__ __forceinline__ int b_k0(int v, int hi) {
  return ((v >> 2) * 32) + hi * 16 + ((v & 3) * 4);
}

__device__ __forceinline__ v16i load_frag16(const unsigned* p) {
  const uint4* q = (const uint4*)p;
  uint4 a = q[0], b = q[1], c = q[2], d = q[3];
  v16i r;
  r[0] = a.x;  r[1] = a.y;  r[2] = a.z;  r[3] = a.w;
  r[4] = b.x;  r[5] = b.y;  r[6] = b.z;  r[7] = b.w;
  r[8] = c.x;  r[9] = c.y;  r[10] = c.z; r[11] = c.w;
  r[12] = d.x; r[13] = d.y; r[14] = d.z; r[15] = d.w;
  return r;
}

__global__ void ffn_prep_kernel(const float* __restrict__ W1, const float* __restrict__ b1,
                                const float* __restrict__ W2, const float* __restrict__ b2,
                                unsigned* __restrict__ ws) {
  int t = blockIdx.x * blockDim.x + threadIdx.x;
  if (t < W1A_N) {
    int ntile = t >> 9, lane = (t >> 4) & 31, v = t & 15;
    int n  = ntile * 16 + (lane & 15);
    int k0 = a_k0(v, (lane >> 4) & 1);
    unsigned w = 0;
#pragma unroll
    for (int j = 0; j < 4; ++j) {
      int k = k0 + j;
      unsigned byte = (k < N_EMBD) ? bf8_byte(W1[n * N_EMBD + k]) : 0u;
      w |= byte << (8 * j);
    }
    ws[W1A_OFF + t] = w;
  } else if (t < W1A_N + W2A_N) {
    int u = t - W1A_N;
    int ktnt = u >> 9, lane = (u >> 4) & 31, v = u & 15;
    int kt = ktnt / NT2, nt = ktnt % NT2;
    int n  = nt * 16 + (lane & 15);
    int k0 = kt * 128 + a_k0(v, (lane >> 4) & 1);
    unsigned w = 0;
#pragma unroll
    for (int j = 0; j < 4; ++j)
      w |= bf8_byte(W2[n * HIDDEN + k0 + j]) << (8 * j);
    ws[W2A_OFF + u] = w;
  } else if (t < W1A_N + W2A_N + HIDDEN) {
    int n = t - (W1A_N + W2A_N);
    ((float*)ws)[B1_OFF + n] = bf8_decode(bf8_byte(b1[n]));
  } else if (t < W1A_N + W2A_N + HIDDEN + N_EMBD) {
    int n = t - (W1A_N + W2A_N + HIDDEN);
    ((float*)ws)[B2_OFF + n] = bf8_decode(bf8_byte(b2[n]));
  }
}

#define WAVES_PER_BLOCK 4

__global__ __launch_bounds__(32 * WAVES_PER_BLOCK)
void ffn_kernel(const float* __restrict__ x, const unsigned* __restrict__ ws,
                float* __restrict__ out, int nTiles) {
  __shared__ unsigned lds_w2[W2A_N];
  __shared__ float    lds_bias[HIDDEN + N_EMBD];

  const int lane = threadIdx.x & 31;
  const int warp = threadIdx.x >> 5;
  const int hi   = (lane >> 4) & 1;
  const int col  = lane & 15;

#if defined(HAVE_ASYNC)
  {
    const unsigned* gsrc = ws + W2A_OFF;
    for (int i = threadIdx.x; i < W2A_N / 4; i += blockDim.x)
      __builtin_amdgcn_global_load_async_to_lds_b128(
          (__attribute__((address_space(1))) v4i*)(gsrc + i * 4),
          (__attribute__((address_space(3))) v4i*)(lds_w2 + i * 4), 0, 0);
    asm volatile("s_wait_asynccnt 0" ::: "memory");
  }
#else
  for (int i = threadIdx.x; i < W2A_N / 4; i += blockDim.x)
    ((uint4*)lds_w2)[i] = ((const uint4*)(ws + W2A_OFF))[i];
#endif
  for (int i = threadIdx.x; i < HIDDEN + N_EMBD; i += blockDim.x)
    lds_bias[i] = ((const float*)ws)[B1_OFF + i];
  __syncthreads();

  const unsigned* W1A = ws + W1A_OFF;
  const float* b1q = lds_bias;
  const float* b2q = lds_bias + HIDDEN;

  const int totalWaves = gridDim.x * WAVES_PER_BLOCK;
  for (int tile = blockIdx.x * WAVES_PER_BLOCK + warp; tile < nTiles; tile += totalWaves) {
    const int token = tile * 16 + col;

    const float* xrow = x + (size_t)token * N_EMBD;
    v16i Bx;
#pragma unroll
    for (int v = 0; v < 16; ++v) {
      const int k0 = b_k0(v, hi);
      if (k0 + 3 < N_EMBD) {
        const float4 f = *(const float4*)(xrow + k0);
        Bx[v] = (int)pack4_bf8(f.x, f.y, f.z, f.w);
      } else {
        Bx[v] = 0;
      }
    }

    const v8f vzero = {0.f, 0.f, 0.f, 0.f, 0.f, 0.f, 0.f, 0.f};
    v8f acc2[NT2];
#pragma unroll
    for (int i = 0; i < NT2; ++i) acc2[i] = vzero;

#pragma unroll 1
    for (int kt = 0; kt < KT2; ++kt) {
      uint2 hq[8];
#pragma unroll
      for (int j = 0; j < 8; ++j) {
        const int nt = kt * 8 + j;
        v16i A = load_frag16(W1A + nt * 512 + lane * 16);
        v8f acc = WMMA128(A, Bx, vzero);
        const float4 ba = *(const float4*)(b1q + nt * 16 + hi * 8);
        const float4 bb = *(const float4*)(b1q + nt * 16 + hi * 8 + 4);
        float hv[8];
#pragma unroll
        for (int d = 0; d < 8; ++d) {
          const float bias = (d < 4) ? ((const float*)&ba)[d] : ((const float*)&bb)[d - 4];
          float h = f16r(f16r(acc[d]) + bias);
          hv[d] = fmaxf(h, 0.0f);
        }
        hq[j].x = pack4_bf8(hv[0], hv[1], hv[2], hv[3]);
        hq[j].y = pack4_bf8(hv[4], hv[5], hv[6], hv[7]);
      }

      v16i Bh;
#pragma unroll
      for (int g = 0; g < 4; ++g) {
        const unsigned w0E = hq[2 * g].x,     w1E = hq[2 * g].y;
        const unsigned w0O = hq[2 * g + 1].x, w1O = hq[2 * g + 1].y;
        const unsigned s0 = (unsigned)__shfl_xor((int)(hi ? w0E : w0O), 16, 32);
        const unsigned s1 = (unsigned)__shfl_xor((int)(hi ? w1E : w1O), 16, 32);
        Bh[4 * g + 0] = (int)(hi ? s0 : w0E);
        Bh[4 * g + 1] = (int)(hi ? s1 : w1E);
        Bh[4 * g + 2] = (int)(hi ? w0O : s0);
        Bh[4 * g + 3] = (int)(hi ? w1O : s1);
      }

#pragma unroll
      for (int nt = 0; nt < NT2; ++nt) {
        v16i A = load_frag16(lds_w2 + (kt * NT2 + nt) * 512 + lane * 16);
        acc2[nt] = WMMA128(A, Bh, acc2[nt]);
      }
    }

#pragma unroll
    for (int nt = 0; nt < NT2; ++nt) {
      const float4 ba = *(const float4*)(b2q + nt * 16 + hi * 8);
      const float4 bb = *(const float4*)(b2q + nt * 16 + hi * 8 + 4);
      float4 o0, o1;
#pragma unroll
      for (int d = 0; d < 4; ++d) {
        ((float*)&o0)[d] = f16r(f16r(acc2[nt][d]) + ((const float*)&ba)[d]);
        ((float*)&o1)[d] = f16r(f16r(acc2[nt][d + 4]) + ((const float*)&bb)[d]);
      }
      float* orow = out + (size_t)token * N_EMBD + nt * 16 + hi * 8;
      *(float4*)orow = o0;
      *(float4*)(orow + 4) = o1;
    }
  }
}

extern "C" void kernel_launch(void* const* d_in, const int* in_sizes, int n_in,
                              void* d_out, int out_size, void* d_ws, size_t ws_size,
                              hipStream_t stream) {
  const float* x  = (const float*)d_in[0];
  const float* W1 = (const float*)d_in[1];
  const float* b1 = (const float*)d_in[2];
  const float* W2 = (const float*)d_in[3];
  const float* b2 = (const float*)d_in[4];
  float* out = (float*)d_out;
  unsigned* ws = (unsigned*)d_ws;

  const int tokens = in_sizes[0] / N_EMBD;
  const int nTiles = tokens / 16;

  const int prepThreads = W1A_N + W2A_N + HIDDEN + N_EMBD;
  ffn_prep_kernel<<<(prepThreads + 255) / 256, 256, 0, stream>>>(W1, b1, W2, b2, ws);

  int blocks = 1024;
  const int needed = (nTiles + WAVES_PER_BLOCK - 1) / WAVES_PER_BLOCK;
  if (blocks > needed) blocks = needed;
  ffn_kernel<<<blocks, 32 * WAVES_PER_BLOCK, 0, stream>>>(x, ws, out, nTiles);
}
